// SimpleTransformerLayer_69140383531292
// MI455X (gfx1250) — matrix operand order repaired and checked by reading, not compiled, not run on hardware
//
#include <hip/hip_runtime.h>
#include <hip/hip_bf16.h>

typedef __attribute__((ext_vector_type(16))) __bf16 v16bf;
typedef __attribute__((ext_vector_type(8)))  float  v8f;

#define DIM_   1024
#define HEADS_ 16
#define DHEAD_ 64
#define INNER_ 1024
#define BATCH_ 2
#define SEQ_   2048
#define ROWS_  4096   // BATCH_*SEQ_
#define DFF_   4096

// ---------- helpers ----------

__device__ __forceinline__ unsigned short f32_to_bf16(float f) {
  unsigned int u = __builtin_bit_cast(unsigned int, f);
  u += 0x7FFFu + ((u >> 16) & 1u);   // round-to-nearest-even
  return (unsigned short)(u >> 16);
}

// CDNA5 async copy: global -> LDS, 16 bytes per lane, tracked by ASYNCcnt.
// LDS byte address = low 32 bits of the generic pointer (flat LDS aperture, ISA 10.2).
__device__ __forceinline__ void async_copy_b128(void* lds_ptr, const void* gptr) {
  unsigned lo = (unsigned)(unsigned long long)lds_ptr;
  asm volatile("global_load_async_to_lds_b128 %0, %1, off"
               :: "v"(lo), "v"(gptr) : "memory");
}
__device__ __forceinline__ void wait_async4() { asm volatile("s_wait_asynccnt 0x4" ::: "memory"); }
__device__ __forceinline__ void wait_async0() { asm volatile("s_wait_asynccnt 0x0" ::: "memory"); }

// A-matrix fragment 16x32 bf16 (MxK), row-major source, contiguous K.
// Lane l (l<16): row M=l, K groups [0..7],[16..23]; lanes 16..31: same rows, K [8..15],[24..31].
__device__ __forceinline__ v16bf load_a_frag(const unsigned short* base, long long stride_m) {
  const int l = threadIdx.x & 31;
  const int m = l & 15;
  const int koff = (l >> 4) * 8;
  const unsigned short* p = base + (long long)m * stride_m + koff;
  union { v16bf f; uint4 q[2]; } u;
  u.q[0] = *(const uint4*)(p);        // K = koff .. koff+7
  u.q[1] = *(const uint4*)(p + 16);   // K = 16+koff .. 23+koff
  return u.f;
}

// B-matrix fragment 32x16 bf16 (KxN) where element (k,n) lives at base[n*stride_n + k]
// (contiguous along K). Lane l: col N=l&15, K groups [(l>>4)*8 ..+7],[16+(l>>4)*8 ..+7].
__device__ __forceinline__ v16bf load_b_frag_t(const unsigned short* base, long long stride_n) {
  const int l = threadIdx.x & 31;
  const int n = l & 15;
  const int koff = (l >> 4) * 8;
  const unsigned short* p = base + (long long)n * stride_n + koff;
  union { v16bf f; uint4 q[2]; } u;
  u.q[0] = *(const uint4*)(p);
  u.q[1] = *(const uint4*)(p + 16);
  return u.f;
}

// ---------- weight convert + transpose: Wt[n*K + k] = bf16(W[k*N + n]) ----------

__global__ __launch_bounds__(256) void convert_transpose_kernel(
    const float* __restrict__ W, unsigned short* __restrict__ Wt, int K, int N) {
  long long idx = (long long)blockIdx.x * 256 + threadIdx.x;
  long long total = (long long)K * N;
  if (idx >= total) return;
  long long k = idx / N, n = idx % N;
  Wt[n * (long long)K + k] = f32_to_bf16(W[idx]);
}

// ---------- V transpose: vt[((b*H+h)*DHEAD+d)*SEQ + s] = V[b,s,h,d] (bf16) ----------

__global__ __launch_bounds__(256) void vtrans_kernel(
    const unsigned short* __restrict__ qkv, unsigned short* __restrict__ vt) {
  long long idx = (long long)blockIdx.x * 256 + threadIdx.x;   // over B*H*DHEAD*SEQ = 4M
  int s = (int)(idx & (SEQ_ - 1));
  long long r = idx >> 11;           // / SEQ_
  int d = (int)(r & (DHEAD_ - 1));
  long long bh = r >> 6;             // / DHEAD_
  int b = (int)(bh / HEADS_);
  int h = (int)(bh % HEADS_);
  vt[idx] = qkv[((long long)(b * SEQ_ + s)) * (3LL * INNER_) + 2 * INNER_ + h * DHEAD_ + d];
}

// ---------- layernorm (f32 in, bf16 out) ----------

__global__ __launch_bounds__(256) void layernorm_bf16_kernel(
    const float* __restrict__ x, const float* __restrict__ g, const float* __restrict__ bta,
    unsigned short* __restrict__ outb) {
  const int row = blockIdx.x;
  const float* xr = x + (long long)row * DIM_;
  __shared__ float s1[256], s2[256];
  float sum = 0.f, sq = 0.f;
  for (int i = threadIdx.x; i < DIM_; i += 256) {
    float v = xr[i]; sum += v; sq += v * v;
  }
  s1[threadIdx.x] = sum; s2[threadIdx.x] = sq;
  __syncthreads();
  for (int s = 128; s > 0; s >>= 1) {
    if (threadIdx.x < s) { s1[threadIdx.x] += s1[threadIdx.x + s]; s2[threadIdx.x] += s2[threadIdx.x + s]; }
    __syncthreads();
  }
  float mu = s1[0] * (1.f / DIM_);
  float var = s2[0] * (1.f / DIM_) - mu * mu;
  float inv = rsqrtf(var + 1e-5f);
  for (int i = threadIdx.x; i < DIM_; i += 256)
    outb[(long long)row * DIM_ + i] = f32_to_bf16((xr[i] - mu) * inv * g[i] + bta[i]);
}

// ---------- bf16 WMMA GEMM with async-LDS double-buffered staging ----------
// C[M,N] = A[M,K] @ B[K,N] (+bias)(+gelu)(+resid).
// A: bf16 row-major [M,K]. Bt: bf16 transposed layout, element (k,n) at Bt[n*K+k].
// Block: 256 thr = 8 waves; wave (wm=w>>1 in 0..3, wn=w&1) computes 32x64.
// Block tile 128(M) x 128(N), k-step 32. LDS: 2 x (8KB A + 8KB B) ping-pong.

__global__ __launch_bounds__(256) void gemm_bf16_kernel(
    const unsigned short* __restrict__ A, const unsigned short* __restrict__ Bt,
    float* __restrict__ Cf, unsigned short* __restrict__ Cb,
    const float* __restrict__ bias, const float* __restrict__ resid,
    int M, int N, int K, int do_gelu) {
  __shared__ __align__(16) unsigned short Ash[2][128][32];
  __shared__ __align__(16) unsigned short Bsh[2][128][32];

  const int w  = threadIdx.x >> 5;
  const int wm = w >> 1;          // 0..3
  const int wn = w & 1;           // 0..1
  const int mBlock = blockIdx.y * 128;
  const int nBlock = blockIdx.x * 128;
  const int tid = threadIdx.x;

  // stage one 128x32 A tile and 128x32 B tile into LDS buffer `buf` (async).
  // 512 16B chunks each; 2 A-chunks + 2 B-chunks per thread = 4 async ops per wave.
  auto stage = [&](int buf, int k0) {
    #pragma unroll
    for (int i = 0; i < 2; ++i) {
      int c = tid * 2 + i;            // 0..511
      int row = c >> 2;
      int kk  = (c & 3) * 8;
      async_copy_b128(&Ash[buf][row][kk],
                      A + (long long)(mBlock + row) * K + k0 + kk);
    }
    #pragma unroll
    for (int i = 0; i < 2; ++i) {
      int c = tid * 2 + i;
      int col = c >> 2;
      int kk  = (c & 3) * 8;
      async_copy_b128(&Bsh[buf][col][kk],
                      Bt + (long long)(nBlock + col) * K + k0 + kk);
    }
  };

  v8f acc[2][4];
  #pragma unroll
  for (int s = 0; s < 2; ++s)
    #pragma unroll
    for (int t = 0; t < 4; ++t)
      #pragma unroll
      for (int r = 0; r < 8; ++r) acc[s][t][r] = 0.f;

  int buf = 0;
  stage(buf, 0);
  for (int k0 = 0; k0 < K; k0 += 32) {
    const bool more = (k0 + 32) < K;
    if (more) stage(buf ^ 1, k0 + 32);    // prefetch next tile while current lands
    if (more) wait_async4(); else wait_async0();   // current tile's 4 copies complete
    __syncthreads();

    const unsigned short* Abase = &Ash[buf][wm * 32][0];
    v16bf a0 = load_a_frag(Abase, 32);
    v16bf a1 = load_a_frag(Abase + 16 * 32, 32);
    const unsigned short* Bbase = &Bsh[buf][wn * 64][0];
    #pragma unroll
    for (int t = 0; t < 4; ++t) {
      v16bf b = load_b_frag_t(Bbase + (t * 16) * 32, 32);
      acc[0][t] = __builtin_amdgcn_wmma_f32_16x16x32_bf16(false, a0, false, b, (short)0, acc[0][t], false, false);
      acc[1][t] = __builtin_amdgcn_wmma_f32_16x16x32_bf16(false, a1, false, b, (short)0, acc[1][t], false, false);
    }
    __syncthreads();                      // all waves done reading before overwrite
    buf ^= 1;
  }

  const int l  = threadIdx.x & 31;
  const int nl = l & 15;
  const int mb = (l >> 4) * 8;
  #pragma unroll
  for (int s = 0; s < 2; ++s) {
    #pragma unroll
    for (int t = 0; t < 4; ++t) {
      #pragma unroll
      for (int r = 0; r < 8; ++r) {
        int m = mBlock + wm * 32 + s * 16 + mb + r;
        int n = nBlock + wn * 64 + t * 16 + nl;
        float v = acc[s][t][r];
        if (bias) v += bias[n];
        if (do_gelu) v = 0.5f * v * (1.f + erff(v * 0.70710678118654752f));
        if (resid) v += resid[(long long)m * N + n];
        if (Cf) Cf[(long long)m * N + n] = v;
        if (Cb) Cb[(long long)m * N + n] = f32_to_bf16(v);
      }
    }
  }
}

// ---------- flash attention, 16 heads, dh=64, bf16 WMMA ----------
// qkv: bf16 [ROWS_, 3*INNER_]; vt: bf16 [B,H,DHEAD,SEQ] (V transposed).
// Each wave: 16 query rows of one (b,h). Per 32-key block: S = Q@K^T (2x2 wmma),
// online softmax, P reshape via per-wave LDS tile, P@V (4 wmma).

__global__ __launch_bounds__(256) void attn_kernel(
    const unsigned short* __restrict__ qkv, const unsigned short* __restrict__ vt,
    unsigned short* __restrict__ attnout) {
  __shared__ __align__(16) unsigned short psh[8][16][32];   // per-wave P tile (16x32 bf16)
  const int w  = threadIdx.x >> 5;
  const int l  = threadIdx.x & 31;
  const int nl = l & 15;
  const int mb = (l >> 4) * 8;
  const int b  = blockIdx.y / HEADS_;
  const int h  = blockIdx.y % HEADS_;
  const int q0 = blockIdx.x * 128 + w * 16;
  const long long LD = 3LL * INNER_;

  const unsigned short* qb = qkv + ((long long)(b * SEQ_ + q0)) * LD + h * DHEAD_;
  v16bf qa0 = load_a_frag(qb, LD);        // d = 0..31
  v16bf qa1 = load_a_frag(qb + 32, LD);   // d = 32..63
  const unsigned short* vtb = vt + ((long long)blockIdx.y * DHEAD_) * SEQ_;

  v8f oacc[4];
  float m_run[8], l_run[8];
  #pragma unroll
  for (int r = 0; r < 8; ++r) { m_run[r] = -3.0e38f; l_run[r] = 0.f; }
  #pragma unroll
  for (int t = 0; t < 4; ++t)
    #pragma unroll
    for (int r = 0; r < 8; ++r) oacc[t][r] = 0.f;

  const float scale = 0.125f;  // dh^-0.5

  for (int kb = 0; kb < SEQ_; kb += 32) {
    // ---- scores: 16x32, two 16-wide N tiles ----
    v8f sc[2];
    #pragma unroll
    for (int t = 0; t < 2; ++t) {
      #pragma unroll
      for (int r = 0; r < 8; ++r) sc[t][r] = 0.f;
      const unsigned short* kp = qkv + ((long long)(b * SEQ_ + kb + t * 16)) * LD + INNER_ + h * DHEAD_;
      v16bf bk0 = load_b_frag_t(kp, LD);        // B[d,key], d contiguous
      v16bf bk1 = load_b_frag_t(kp + 32, LD);
      sc[t] = __builtin_amdgcn_wmma_f32_16x16x32_bf16(false, qa0, false, bk0, (short)0, sc[t], false, false);
      sc[t] = __builtin_amdgcn_wmma_f32_16x16x32_bf16(false, qa1, false, bk1, (short)0, sc[t], false, false);
    }
    // ---- online softmax over this 32-key block ----
    float p0a[8], p1a[8], corr[8];
    #pragma unroll
    for (int r = 0; r < 8; ++r) {
      float s0 = sc[0][r] * scale;
      float s1 = sc[1][r] * scale;
      float mx = fmaxf(s0, s1);
      #pragma unroll
      for (int d = 1; d < 16; d <<= 1) mx = fmaxf(mx, __shfl_xor(mx, d, 32));
      float mnew = fmaxf(m_run[r], mx);
      corr[r] = __expf(m_run[r] - mnew);
      float p0 = __expf(s0 - mnew);
      float p1 = __expf(s1 - mnew);
      float ps = p0 + p1;
      #pragma unroll
      for (int d = 1; d < 16; d <<= 1) ps += __shfl_xor(ps, d, 32);
      l_run[r] = l_run[r] * corr[r] + ps;
      m_run[r] = mnew;
      p0a[r] = p0; p1a[r] = p1;
    }
    #pragma unroll
    for (int t = 0; t < 4; ++t)
      #pragma unroll
      for (int r = 0; r < 8; ++r) oacc[t][r] *= corr[r];

    // ---- reshape P (C-layout -> A-layout) through per-wave LDS tile ----
    #pragma unroll
    for (int r = 0; r < 8; ++r) {
      psh[w][mb + r][nl]      = f32_to_bf16(p0a[r]);
      psh[w][mb + r][16 + nl] = f32_to_bf16(p1a[r]);
    }
    asm volatile("s_wait_dscnt 0x0" ::: "memory");   // wave-private tile; DS in-order per wave
    v16bf pa = load_a_frag(&psh[w][0][0], 32);

    // ---- out += P @ V  (V^T layout: element (key,d) at vtb[d*SEQ + kb + key]) ----
    #pragma unroll
    for (int t = 0; t < 4; ++t) {
      v16bf bv = load_b_frag_t(vtb + (long long)(t * 16) * SEQ_ + kb, SEQ_);
      oacc[t] = __builtin_amdgcn_wmma_f32_16x16x32_bf16(false, pa, false, bv, (short)0, oacc[t], false, false);
    }
  }

  #pragma unroll
  for (int t = 0; t < 4; ++t) {
    #pragma unroll
    for (int r = 0; r < 8; ++r) {
      int q = q0 + mb + r;
      float o = oacc[t][r] / l_run[r];
      attnout[((long long)(b * SEQ_ + q)) * INNER_ + h * DHEAD_ + t * 16 + nl] = f32_to_bf16(o);
    }
  }
}

// ---------- host launcher ----------

extern "C" void kernel_launch(void* const* d_in, const int* in_sizes, int n_in,
                              void* d_out, int out_size, void* d_ws, size_t ws_size,
                              hipStream_t stream) {
  const float* x     = (const float*)d_in[0];
  const float* w_qkv = (const float*)d_in[1];
  const float* w_out = (const float*)d_in[2];
  const float* b_out = (const float*)d_in[3];
  const float* g1    = (const float*)d_in[4];
  const float* beta1 = (const float*)d_in[5];
  const float* g2    = (const float*)d_in[6];
  const float* beta2 = (const float*)d_in[7];
  const float* w_ff1 = (const float*)d_in[8];
  const float* b_ff1 = (const float*)d_in[9];
  const float* w_ff2 = (const float*)d_in[10];
  const float* b_ff2 = (const float*)d_in[11];
  float* out = (float*)d_out;

  char* ws = (char*)d_ws;
  size_t off = 0;
  auto alloc = [&](size_t bytes) -> char* {
    char* p = ws + off;
    off += (bytes + 255) & ~(size_t)255;
    return p;
  };
  unsigned short* wqkv_t = (unsigned short*)alloc((size_t)DIM_ * 3 * INNER_ * 2);  // 6 MB
  unsigned short* wout_t = (unsigned short*)alloc((size_t)INNER_ * DIM_ * 2);      // 2 MB
  unsigned short* wff1_t = (unsigned short*)alloc((size_t)DIM_ * DFF_ * 2);        // 8 MB
  unsigned short* wff2_t = (unsigned short*)alloc((size_t)DFF_ * DIM_ * 2);        // 8 MB
  unsigned short* hbuf   = (unsigned short*)alloc((size_t)ROWS_ * DIM_ * 2);       // 8 MB (h1 & h2)
  char*           big    = alloc((size_t)ROWS_ * 3 * INNER_ * 2 + (size_t)ROWS_ * INNER_ * 2); // 32 MB
  unsigned short* vtb    = (unsigned short*)alloc((size_t)BATCH_ * HEADS_ * DHEAD_ * SEQ_ * 2); // 8 MB
  float*          x_attn = (float*)alloc((size_t)ROWS_ * DIM_ * 4);                // 16 MB

  unsigned short* qkvb     = (unsigned short*)big;
  unsigned short* attnoutb = (unsigned short*)(big + (size_t)ROWS_ * 3 * INNER_ * 2);
  unsigned short* ffb      = (unsigned short*)big;  // [ROWS_, DFF_] reuses dead qkv+attnout

  // 1) convert+transpose weights to bf16 [N,K]
  {
    long long e;
    e = (long long)DIM_ * 3 * INNER_;
    convert_transpose_kernel<<<dim3((unsigned)((e + 255) / 256)), 256, 0, stream>>>(w_qkv, wqkv_t, DIM_, 3 * INNER_);
    e = (long long)INNER_ * DIM_;
    convert_transpose_kernel<<<dim3((unsigned)((e + 255) / 256)), 256, 0, stream>>>(w_out, wout_t, INNER_, DIM_);
    e = (long long)DIM_ * DFF_;
    convert_transpose_kernel<<<dim3((unsigned)((e + 255) / 256)), 256, 0, stream>>>(w_ff1, wff1_t, DIM_, DFF_);
    e = (long long)DFF_ * DIM_;
    convert_transpose_kernel<<<dim3((unsigned)((e + 255) / 256)), 256, 0, stream>>>(w_ff2, wff2_t, DFF_, DIM_);
  }

  // 2) LN1 -> bf16
  layernorm_bf16_kernel<<<dim3(ROWS_), 256, 0, stream>>>(x, g1, beta1, hbuf);

  // 3) qkv = h @ w_qkv  (bf16 out only)
  gemm_bf16_kernel<<<dim3(3 * INNER_ / 128, ROWS_ / 128), 256, 0, stream>>>(
      hbuf, wqkv_t, nullptr, qkvb, nullptr, nullptr, ROWS_, 3 * INNER_, DIM_, 0);

  // 3b) V transpose for attention
  vtrans_kernel<<<dim3((BATCH_ * HEADS_ * DHEAD_ * SEQ_) / 256), 256, 0, stream>>>(qkvb, vtb);

  // 4) flash attention
  attn_kernel<<<dim3(SEQ_ / 128, BATCH_ * HEADS_), 256, 0, stream>>>(qkvb, vtb, attnoutb);

  // 5) x_attn = attnout @ w_out + b_out + x  (f32 out)
  gemm_bf16_kernel<<<dim3(DIM_ / 128, ROWS_ / 128), 256, 0, stream>>>(
      attnoutb, wout_t, x_attn, nullptr, b_out, x, ROWS_, DIM_, INNER_, 0);

  // 6) LN2 -> bf16
  layernorm_bf16_kernel<<<dim3(ROWS_), 256, 0, stream>>>(x_attn, g2, beta2, hbuf);

  // 7) ff = gelu(h2 @ w_ff1 + b_ff1)  (bf16 out, reuses qkv space)
  gemm_bf16_kernel<<<dim3(DFF_ / 128, ROWS_ / 128), 256, 0, stream>>>(
      hbuf, wff1_t, nullptr, ffb, b_ff1, nullptr, ROWS_, DFF_, DIM_, 1);

  // 8) out = ff @ w_ff2 + b_ff2 + x_attn  (f32 out)
  gemm_bf16_kernel<<<dim3(DIM_ / 128, ROWS_ / 128), 256, 0, stream>>>(
      ffb, wff2_t, out, nullptr, b_ff2, x_attn, ROWS_, DIM_, DFF_, 0);

  (void)in_sizes; (void)n_in; (void)out_size; (void)ws_size;
}
